// CgpHmmCell_20126216749373
// MI455X (gfx1250) — matrix operand order repaired and checked by reading, not compiled, not run on hardware
//
#include <hip/hip_runtime.h>

#define NSTATE 108
#define SPAD   128
#define NEMIT  126
#define TLEN   2048
#define NC     16

typedef __attribute__((ext_vector_type(16))) __bf16 v16bf;
typedef __attribute__((ext_vector_type(8)))  float  v8f;

static __device__ __forceinline__ unsigned short f2bf(float f) {
  unsigned u = __float_as_uint(f);
  unsigned r = ((u >> 16) & 1u) + 0x7FFFu;   // round-to-nearest-even
  return (unsigned short)((u + r) >> 16);
}

// VALU 16-lane sum via DPP (no DS traffic): xor1, xor2, cross-4, cross-8
#define DPP_ADD(x, ctrl) \
  ((x) + __int_as_float(__builtin_amdgcn_mov_dpp(__float_as_int(x), (ctrl), 0xf, 0xf, true)))
static __device__ __forceinline__ float sum16_dpp(float x) {
  x = DPP_ADD(x, 0xB1);    // quad_perm(1,0,3,2)  : lane ^ 1
  x = DPP_ADD(x, 0x4E);    // quad_perm(2,3,0,1)  : lane ^ 2
  x = DPP_ADD(x, 0x141);   // ROW_HALF_MIRROR     : combine 4-groups within 8
  x = DPP_ADD(x, 0x140);   // ROW_MIRROR          : combine 8-groups within 16
  return x;
}

// ---------------- emission-structure replication ----------------
#define SY_A 0
#define SY_C 1
#define SY_G 2
#define SY_T 3
#define SY_N 4

static __device__ __forceinline__ int symMask(int s) {
  switch (s) {
    case SY_A: return 1 << 0;
    case SY_C: return 1 << 1;
    case SY_G: return 1 << 2;
    case SY_T: return 1 << 3;
    default:   return 0xF;        // N = {A,C,G,T}
  }
}

struct CallT { int state, s0, s1, s2, xmust; bool train, isX; };

static __device__ CallT getCall(int ci) {
  CallT c; c.xmust = 2; c.train = true; c.isX = false;
  c.s0 = SY_N; c.s1 = SY_N; c.s2 = SY_N;
  switch (ci) {
    case 0:  c.state = 0;  c.xmust = 0; return c;                                   // "N"
    case 1:  c.state = 1;  c.s2 = SY_A; c.xmust = 0; return c;                      // "A"
    case 2:  c.state = 2;  c.s1 = SY_A; c.s2 = SY_T; c.xmust = 0; return c;         // "AT"
    case 3:  c.state = 3;  c.s0 = SY_A; c.s1 = SY_T; c.s2 = SY_G; c.train = false; return c; // "ATG"
    case 4:  c.state = 4;  c.s0 = SY_T; c.s1 = SY_G; c.s2 = SY_N; return c;         // "ATGN" -> TGN
    case 5:  c.state = 5;  c.s0 = SY_G; return c;                                   // "ATGNN" -> GNN
    case 52: c.state = 52; c.s2 = SY_T; return c;                                   // "T"
    case 53: c.state = 53; c.s1 = SY_T; c.s2 = SY_A; return c;                      // "TA"
    case 54: c.state = 53; c.s1 = SY_T; c.s2 = SY_G; return c;                      // "TG"
    case 55: c.state = 54; c.s0 = SY_T; c.s1 = SY_A; c.s2 = SY_A; c.train = false; return c; // "TAA"
    case 56: c.state = 54; c.s0 = SY_T; c.s1 = SY_A; c.s2 = SY_G; c.train = false; return c; // "TAG"
    case 57: c.state = 54; c.s0 = SY_T; c.s1 = SY_G; c.s2 = SY_A; c.train = false; return c; // "TGA"
    case 110: c.state = 107; c.isX = true; return c;                                // "X"
    default:
      c.state = (ci <= 51) ? ci : (ci - 3);  // runs of "N" states
      return c;
  }
}

// Enumerate one call's (col) set in the reference's product order.
// If Blog != nullptr, scatter logits; returns #trainable entries consumed.
static __device__ int enumCall(const CallT& c, const float* ek, int kbase, float* Blog) {
  int m0 = symMask(c.s0) | ((0 < 2 - c.xmust) ? 16 : 0);
  int m1 = symMask(c.s1) | ((1 < 2 - c.xmust) ? 16 : 0);
  int m2 = symMask(c.s2);
  int k = 0;
  for (int x0 = 0; x0 < 6; ++x0) {
    if (!((m0 >> x0) & 1)) continue;
    for (int x1 = 0; x1 < 6; ++x1) {
      if (!((m1 >> x1) & 1)) continue;
      if (x0 != 4 && x1 == 4) continue;          // "found" rule
      for (int x2 = 0; x2 < 6; ++x2) {
        if (!((m2 >> x2) & 1)) continue;
        int col = x0 * 25 + x1 * 5 + x2;
        if (Blog) Blog[c.state * NEMIT + col] = c.train ? ek[kbase + k] : 1.0f;
        if (c.train) ++k;
      }
    }
  }
  return k;
}

// ---------------- build kernel: A (swizzled bf16), Bt (f32), pi ----------------
__global__ void build_kernel(const float* __restrict__ w, const float* __restrict__ ek,
                             const float* __restrict__ initk,
                             float* __restrict__ pi, unsigned short* __restrict__ Asw,
                             float* __restrict__ Bt, float* __restrict__ Alog,
                             float* __restrict__ Blog) {
  const int tid = threadIdx.x;
  __shared__ int cnt[112], kbase[112];

  for (int i = tid; i < SPAD * SPAD;   i += 256) Alog[i] = -1e30f;
  for (int i = tid; i < NSTATE * NEMIT; i += 256) Blog[i] = -1e30f;
  for (int i = tid; i < NEMIT * SPAD;  i += 256) Bt[i] = 0.f;
  __syncthreads();

  if (tid == 0) {  // _build_A scatter (tiny, serial)
    const int n = NC, off = 8 + 3 * NC, term1 = 11 + 6 * NC;
    int k;
#define PUT(r, c, v) Alog[(r) * SPAD + (c)] = (v)
    PUT(0, 0, 1.f - w[0]); PUT(0, 1, w[0]); PUT(1, 2, 1.f); PUT(2, 3, 1.f);
    k = 1;
    for (int i = 0; i < n; ++i) PUT(3 + 3 * i, 4 + 3 * i, w[k + i]);
    k += n;
    for (int i = 0; i < n; ++i) { PUT(4 + 3 * i, 5 + 3 * i, 1.f); PUT(5 + 3 * i, 6 + 3 * i, 1.f); }
    for (int i = 0; i <= n; ++i) PUT(3 + 3 * i, off + 3 * i, w[k + i]);
    k += n + 1;
    PUT(3 + 3 * n, 4 + 3 * n, w[k]); k += 1;
    for (int i = 0; i <= n; ++i) { PUT(off + 3 * i, off + 1 + 3 * i, 1.f); PUT(off + 1 + 3 * i, off + 2 + 3 * i, 1.f); }
    for (int i = 0; i <= n; ++i) { PUT(off + 2 + 3 * i, 4 + 3 * i, w[k + i]); PUT(off + 2 + 3 * i, off + 3 * i, 1.f - w[k + i]); }
    k += n + 1;
    float wk = w[k];
    for (int i = 0; i < n; ++i)
      for (int j = i + 1; j <= n; ++j) {
        int e = j - i + 1; float p = 1.f;
        for (int q = 0; q < e; ++q) p *= wk;   // wk**e, sign handled exactly
        PUT(3 + 3 * i, 4 + 3 * j, 1.f - p);
      }
    PUT(4 + 3 * n, 5 + 3 * n, 1.f); PUT(5 + 3 * n, 6 + 3 * n, 1.f); PUT(6 + 3 * n, 7 + 3 * n, 1.f);
    PUT(7 + 3 * n, 7 + 3 * n, 1.f); PUT(7 + 3 * n, term1, 1.f); PUT(term1, term1, 1.f);
#undef PUT
  }

  if (tid < 111) {  // per-call trainable-entry counts
    CallT c = getCall(tid);
    cnt[tid] = c.isX ? 0 : enumCall(c, nullptr, 0, nullptr);
  }
  __syncthreads();
  if (tid == 0) { int a = 0; for (int i = 0; i < 111; ++i) { kbase[i] = a; a += cnt[i]; } }
  __syncthreads();
  if (tid < 111) {  // scatter B logits
    CallT c = getCall(tid);
    if (c.isX) Blog[107 * NEMIT + 125] = 1.0f;
    else enumCall(c, ek, kbase[tid], Blog);
  }
  __syncthreads();

  if (tid < NSTATE) {            // B row-softmax -> Bt[col][state]
    int r = tid;
    float m = -1e30f;
    for (int c = 0; c < NEMIT; ++c) m = fmaxf(m, Blog[r * NEMIT + c]);
    if (m > -1e29f) {
      float z = 0.f;
      for (int c = 0; c < NEMIT; ++c) z += expf(Blog[r * NEMIT + c] - m);
      float iz = 1.f / z;
      for (int c = 0; c < NEMIT; ++c) {
        float v = Blog[r * NEMIT + c];
        if (v > -1e29f) Bt[c * SPAD + r] = expf(v - m) * iz;
      }
    }
  }
  if (tid >= 128) {              // A row-softmax (in place), pad rows -> 0
    int r = tid - 128;
    float m = -1e30f;
    for (int c = 0; c < SPAD; ++c) m = fmaxf(m, Alog[r * SPAD + c]);
    if (m > -1e29f) {
      float z = 0.f;
      for (int c = 0; c < SPAD; ++c) z += expf(Alog[r * SPAD + c] - m);
      float iz = 1.f / z;
      for (int c = 0; c < SPAD; ++c) {
        float v = Alog[r * SPAD + c];
        Alog[r * SPAD + c] = (v > -1e29f) ? expf(v - m) * iz : 0.f;
      }
    } else {
      for (int c = 0; c < SPAD; ++c) Alog[r * SPAD + c] = 0.f;
    }
  }
  if (tid == 0) {                // pi = softmax(init_kernel), padded with 0
    float m = -1e30f;
    for (int i = 0; i < NSTATE; ++i) m = fmaxf(m, initk[i]);
    float z = 0.f;
    for (int i = 0; i < NSTATE; ++i) z += expf(initk[i] - m);
    float iz = 1.f / z;
    for (int i = 0; i < NSTATE; ++i) pi[i] = expf(initk[i] - m) * iz;
    for (int i = NSTATE; i < SPAD; ++i) pi[i] = 0.f;
  }
  __syncthreads();

  // swizzle A into WMMA B-operand layout: [ntile][kchunk][lane][elem]
  for (int idx = tid; idx < 8 * 4 * 32 * 16; idx += 256) {
    int e  = idx & 15;
    int l  = (idx >> 4) & 31;
    int kc = (idx >> 9) & 3;
    int wv = idx >> 11;
    int K   = kc * 32 + ((e >> 3) << 4) + ((l >> 4) << 3) + (e & 7);   // table order: k = 8h+s (s<8), 16+8h+(s-8) (s>=8)
    int col = wv * 16 + (l & 15);
    Asw[idx] = f2bf(Alog[K * SPAD + col]);
  }
}

// ---------------- main forward-scan kernel ----------------
__global__ __launch_bounds__(256, 1)
void hmm_fwd(const int* __restrict__ inputs, const float* __restrict__ gPi,
             const unsigned short* __restrict__ gAsw, const float* __restrict__ gBt,
             float* __restrict__ out) {
  __shared__ float sBt[NEMIT * SPAD];                            // emissions f32 (64.5KB)
  __shared__ __align__(32) __bf16 sAl[2][4 * 32 * 16];           // alpha bf16, A-operand layout, 2x buf
  __shared__ __align__(16) float sPart[16][8];                   // row-sum partials, [m][wave]
  __shared__ __align__(16) int   sEmit[128][16];                 // staged emission ids

  const int tid  = threadIdx.x, bt = blockIdx.x;
  const int wave = tid >> 5, lane = tid & 31, half = lane >> 4, nn = lane & 15;
  const int colBase = wave * 16 + nn;                            // this lane's output column
  // A-operand stash coordinates for column colBase (K index = colBase)
  const int skc = colBase >> 5, skk = colBase & 31;
  const int sel = ((skk >> 4) << 3) | (((skk >> 1) & 3) << 1) | (skk & 1);
  const int sh2 = (skk >> 3) & 1;

  for (int i = tid; i < NEMIT * SPAD; i += 256) sBt[i] = gBt[i];
  {  // stage emission chunk 0 (16 rows x 128 steps), coalesced per row
    int row = tid >> 4, j = tid & 15;
    const int* src = inputs + (size_t)(bt * 16 + row) * TLEN + j * 8;
#pragma unroll
    for (int q = 0; q < 8; ++q) sEmit[j * 8 + q][row] = src[q];
  }
  // hoist this wave's A-matrix tile (constant B operands) into registers
  v16bf bv0 = *(const v16bf*)(const void*)(gAsw + ((wave * 4 + 0) * 32 + lane) * 16);
  v16bf bv1 = *(const v16bf*)(const void*)(gAsw + ((wave * 4 + 1) * 32 + lane) * 16);
  v16bf bv2 = *(const v16bf*)(const void*)(gAsw + ((wave * 4 + 2) * 32 + lane) * 16);
  v16bf bv3 = *(const v16bf*)(const void*)(gAsw + ((wave * 4 + 3) * 32 + lane) * 16);
  const float piv = gPi[colBase];
  __syncthreads();

  int eids[8];                                   // loop-carried emission ids (this step)
  auto loadEids = [&](int te) {
    int4 ea = *(const int4*)&sEmit[te][half * 8];
    int4 eb = *(const int4*)&sEmit[te][half * 8 + 4];
    eids[0] = ea.x; eids[1] = ea.y; eids[2] = ea.z; eids[3] = ea.w;
    eids[4] = eb.x; eids[5] = eb.y; eids[6] = eb.z; eids[7] = eb.w;
  };

  auto tail = [&](float (&val)[8], int t) {
    float s[8];
#pragma unroll
    for (int r = 0; r < 8; ++r) s[r] = sum16_dpp(val[r]);   // 16-lane sums, pure VALU
    if (nn == 0) {
#pragma unroll
      for (int r = 0; r < 8; ++r) sPart[half * 8 + r][wave] = s[r];
    }
    __syncthreads();                // barrier 1: partials ready
    const int wb = (t & 1) ^ 1;
#pragma unroll
    for (int r = 0; r < 8; ++r) {
      int m = half * 8 + r;
      const float4* pp = (const float4*)sPart[m];
      float4 x = pp[0], y = pp[1];
      float sum = ((x.x + x.y) + (x.z + x.w)) + ((y.x + y.y) + (y.z + y.w));
      float v = val[r] * __builtin_amdgcn_rcpf(sum);
      if (colBase < NSTATE)
        out[(size_t)(bt * 16 + m) * ((size_t)TLEN * NSTATE) + (size_t)t * NSTATE + colBase] = v;
      sAl[wb][(skc * 32 + (sh2 * 16 + m)) * 16 + sel] = (__bf16)v;   // stash for next step
    }
    // barrier 2, split: prefetch next step's emission ids in the signal->wait shadow
    asm volatile("s_wait_dscnt 0x0\n\ts_barrier_signal -1" ::: "memory");
    const int tn = t + 1;
    if (tn < TLEN && (tn & 127) != 0) loadEids(tn & 127);
    asm volatile("s_barrier_wait -1" ::: "memory");
  };

  {  // t = 0 : a0 = pi * Bt[e0], normalized
    loadEids(0);
    float val[8];
#pragma unroll
    for (int r = 0; r < 8; ++r)
      val[r] = piv * sBt[eids[r] * SPAD + colBase];
    tail(val, 0);
  }

  for (int t = 1; t < TLEN; ++t) {
    if ((t & 127) == 0) {  // restage 128 emission steps
      int row = tid >> 4, j = tid & 15;
      const int* src = inputs + (size_t)(bt * 16 + row) * TLEN + t + j * 8;
#pragma unroll
      for (int q = 0; q < 8; ++q) sEmit[j * 8 + q][row] = src[q];
      __syncthreads();
      loadEids(0);
    }
    const int rb = t & 1;
    // alpha[16,128] @ A[128, 16-col tile] as two independent WMMA chains
    v16bf a0 = *(const v16bf*)(const void*)&sAl[rb][(0 * 32 + lane) * 16];
    v16bf a1 = *(const v16bf*)(const void*)&sAl[rb][(1 * 32 + lane) * 16];
    v16bf a2 = *(const v16bf*)(const void*)&sAl[rb][(2 * 32 + lane) * 16];
    v16bf a3 = *(const v16bf*)(const void*)&sAl[rb][(3 * 32 + lane) * 16];
    v8f acc0 = {}, acc1 = {};
    acc0 = __builtin_amdgcn_wmma_f32_16x16x32_bf16(false, a0, false, bv0, (short)0, acc0, false, false);
    acc1 = __builtin_amdgcn_wmma_f32_16x16x32_bf16(false, a1, false, bv1, (short)0, acc1, false, false);
    acc0 = __builtin_amdgcn_wmma_f32_16x16x32_bf16(false, a2, false, bv2, (short)0, acc0, false, false);
    acc1 = __builtin_amdgcn_wmma_f32_16x16x32_bf16(false, a3, false, bv3, (short)0, acc1, false, false);

    float val[8];
#pragma unroll
    for (int r = 0; r < 8; ++r)      // emission multiply (per-row gathered Bt row)
      val[r] = (acc0[r] + acc1[r]) * sBt[eids[r] * SPAD + colBase];
    tail(val, t);
  }
}

extern "C" void kernel_launch(void* const* d_in, const int* in_sizes, int n_in,
                              void* d_out, int out_size, void* d_ws, size_t ws_size,
                              hipStream_t stream) {
  (void)in_sizes; (void)n_in; (void)out_size; (void)ws_size;
  const int*   inputs = (const int*)d_in[0];
  const float* w      = (const float*)d_in[1];
  const float* ek     = (const float*)d_in[2];
  const float* initk  = (const float*)d_in[3];
  float* out = (float*)d_out;

  char* ws = (char*)d_ws;
  float*          pi   = (float*)(ws);                                   // 512 B
  unsigned short* Asw  = (unsigned short*)(ws + 512);                    // 32768 B
  float*          Bt   = (float*)(ws + 512 + 32768);                     // 64512 B
  float*          Alog = (float*)(ws + 512 + 32768 + 64512);             // 65536 B
  float*          Blog = (float*)(ws + 512 + 32768 + 64512 + 65536);     // 54432 B

  hipLaunchKernelGGL(build_kernel, dim3(1), dim3(256), 0, stream, w, ek, initk, pi, Asw, Bt, Alog, Blog);
  hipLaunchKernelGGL(hmm_fwd, dim3(16), dim3(256), 0, stream, inputs, pi, Asw, Bt, out);
}
